// MultiHeadAttention_68771016343792
// MI455X (gfx1250) — matrix operand order repaired and checked by reading, not compiled, not run on hardware
//
#include <hip/hip_runtime.h>
#include <hip/hip_bf16.h>

// ---- problem constants (match reference) ----
#define BB   2
#define SEQ  2048
#define NH   16
#define DH   64
#define DEMB (NH*DH)

typedef __attribute__((ext_vector_type(16))) _Float16 v16h;
typedef __attribute__((ext_vector_type(8)))  _Float16 v8h;
typedef __attribute__((ext_vector_type(8)))  float    v8f;

// D = A(16x32 f16) * B(32x16 f16) + C(16x16 f32), wave32 WMMA
__device__ __forceinline__ v8f wmma16(v16h a, v16h b, v8f c) {
  return __builtin_amdgcn_wmma_f32_16x16x32_f16(false, a, false, b, (short)0, c,
                                                false, false);
}

// A-operand fragment (16x32, f16) from a 64-element f16 row in memory.
// Lane layout: row m = lane&15; elements 0..15 -> K = g*16+e,
// g = lane>>4. fragoff selects K base (0 or 32).
__device__ __forceinline__ v16h load_a_h16(const _Float16* row, int g, int fragoff) {
  const _Float16* p = row + fragoff + g * 16;
  v8h lo = *(const v8h*)(p);
  v8h hi = *(const v8h*)(p + 8);
  v16h a;
#pragma unroll
  for (int e = 0; e < 8; ++e) { a[e] = lo[e]; a[8 + e] = hi[e]; }
  return a;
}

// ============================================================================
// Kernel 1: per-head projections  y = x @ W^T  (torch Linear convention),
// f32 -> f16, via WMMA.  Writes:
//   Qp[b][h][s][d] (pre-scaled by log2(e)/sqrt(Dh) for exp2-domain softmax),
//   Kp[b][h][s][d],
//   Vt[b][h][d][s] (transposed so attention's PV B-operand is contiguous).
// One wave handles (which, b, h, 16-row s-tile).
// ============================================================================
__global__ void mha_proj_kernel(const float* __restrict__ qin,
                                const float* __restrict__ kin,
                                const float* __restrict__ vin,
                                const float* __restrict__ Wq,
                                const float* __restrict__ Wk,
                                const float* __restrict__ Wv,
                                _Float16* __restrict__ Qp,
                                _Float16* __restrict__ Kp,
                                _Float16* __restrict__ Vt) {
  const int STILES = SEQ / 16;
  int wave = (blockIdx.x * blockDim.x + threadIdx.x) >> 5;
  int lane = threadIdx.x & 31;
  int g = lane >> 4, ln = lane & 15;

  int st = wave % STILES; wave /= STILES;
  int h  = wave % NH;     wave /= NH;
  int b  = wave % BB;     wave /= BB;
  int which = wave;  // 0=q, 1=k, 2=v

  const float* x = (which == 0) ? qin : (which == 1) ? kin : vin;
  const float* W = (which == 0) ? Wq  : (which == 1) ? Wk  : Wv;

  // A fragments: 16 s-rows x 64 input dims (f32 -> f16)
  const float* xrow = x + ((size_t)b * SEQ + st * 16 + ln) * DEMB + h * DH;
  v16h a0, a1;
#pragma unroll
  for (int e = 0; e < 8; ++e) {
    a0[e]     = (_Float16)xrow[g * 16 + e];
    a0[8 + e] = (_Float16)xrow[g * 16 + 8 + e];
    a1[e]     = (_Float16)xrow[32 + g * 16 + e];
    a1[8 + e] = (_Float16)xrow[32 + g * 16 + 8 + e];
  }

  // fold 1/sqrt(64) AND log2(e) into Q so attention uses raw v_exp_f32 (base 2)
  const float scl = (which == 0) ? (0.125f * 1.44269504088896f) : 1.0f;

#pragma unroll
  for (int t = 0; t < 4; ++t) {  // 4 output-dim tiles of 16
    // B operand: B[k=e_in][n=d_out] = W[d_out][e_in]; lane: n = t*16+ln,
    // k = f*32 + g*16 + e  (contiguous along e_in -> contiguous in W row)
    int d = t * 16 + ln;
    const float* wp0 = W + d * DH + g * 16;
    const float* wp1 = W + d * DH + 32 + g * 16;
    v16h b0, b1;
#pragma unroll
    for (int e = 0; e < 16; ++e) { b0[e] = (_Float16)wp0[e]; b1[e] = (_Float16)wp1[e]; }

    v8f c = {};
    c = wmma16(a0, b0, c);
    c = wmma16(a1, b1, c);

    if (which == 2) {
      // Vt[((b*NH+h)*DH + d)*SEQ + pos]; key st*16+g*8+r at pos (st>>1)*32 + g*16 + (st&1)*8 + r -> one 16B store
      _Float16* vp = Vt + ((size_t)(b * NH + h) * DH + d) * SEQ + (st >> 1) * 32 + g * 16 + (st & 1) * 8;
      v8h pk;
#pragma unroll
      for (int r = 0; r < 8; ++r) pk[r] = (_Float16)c[r];
      *(v8h*)vp = pk;
    } else {
      _Float16* op = ((which == 0) ? Qp : Kp) +
                     ((size_t)(b * NH + h) * SEQ + st * 16 + g * 8) * DH + d;
#pragma unroll
      for (int r = 0; r < 8; ++r) op[(size_t)r * DH] = (_Float16)(c[r] * scl);
    }
  }
}

// ============================================================================
// Kernel 2: flash attention.  One wave owns (b, h, 32-query block = 2 WMMA
// q-tiles) and streams 32 keys/iteration; K/V fragments are loaded once and
// reused for both q-tiles (halves L2 traffic vs 16-query waves).
// Computes S^T = K_tile x Q^T so the score C-layout IS the P A-layout for
// P x V (no transpose needed).  Online softmax in fp32, exp2 domain.
// ============================================================================
__global__ void mha_attn_kernel(const _Float16* __restrict__ Qp,
                                const _Float16* __restrict__ Kp,
                                const _Float16* __restrict__ Vt,
                                float* __restrict__ out) {
  const int QB = SEQ / 32;  // 32-query blocks per (b,h)
  int wave = (blockIdx.x * blockDim.x + threadIdx.x) >> 5;
  int lane = threadIdx.x & 31;
  int g = lane >> 4, ln = lane & 15;

  int qb = wave % QB; wave /= QB;
  int h  = wave % NH; wave /= NH;
  int b  = wave;
  int qbase = qb * 32;

  // Q^T B-operand fragments for both q-tiles: lane: n = query = ln,
  // k = d = g*16 + e  (frag 0: d 0..31, frag 1: d 32..63)
  const _Float16* qrowbase = Qp + (size_t)(b * NH + h) * SEQ * DH;
  const _Float16* qr0 = qrowbase + (size_t)(qbase + ln) * DH;
  const _Float16* qr1 = qrowbase + (size_t)(qbase + 16 + ln) * DH;
  v16h qb00 = *(const v16h*)(qr0 + g * 16);
  v16h qb01 = *(const v16h*)(qr0 + 32 + g * 16);
  v16h qb10 = *(const v16h*)(qr1 + g * 16);
  v16h qb11 = *(const v16h*)(qr1 + 32 + g * 16);

  const _Float16* kbase = Kp + (size_t)(b * NH + h) * SEQ * DH;
  const _Float16* vbase = Vt + (size_t)(b * NH + h) * DH * SEQ;

  v8f o00 = {}, o01 = {}, o02 = {}, o03 = {};  // q-tile 0, 4 d-tiles
  v8f o10 = {}, o11 = {}, o12 = {}, o13 = {};  // q-tile 1
  float m0 = -__builtin_inff(), l0 = 0.0f;
  float m1 = -__builtin_inff(), l1 = 0.0f;

  for (int it = 0; it < SEQ / 32; ++it) {
    int kv = it * 32;

    // ---- shared K A-frags (32 keys x 64 d) and V B-frags (32 keys x 64 d) ----
    const _Float16* kr0 = kbase + (size_t)(kv + ln) * DH;        // keys 0..15
    const _Float16* kr1 = kbase + (size_t)(kv + 16 + ln) * DH;   // keys 16..31
    v16h ka00 = load_a_h16(kr0, g, 0);
    v16h ka01 = load_a_h16(kr0, g, 32);
    v16h ka10 = load_a_h16(kr1, g, 0);
    v16h ka11 = load_a_h16(kr1, g, 32);

    v16h vb0 = *(const v16h*)(vbase + (size_t)(0 * 16 + ln) * SEQ + kv + g * 16);
    v16h vb1 = *(const v16h*)(vbase + (size_t)(1 * 16 + ln) * SEQ + kv + g * 16);
    v16h vb2 = *(const v16h*)(vbase + (size_t)(2 * 16 + ln) * SEQ + kv + g * 16);
    v16h vb3 = *(const v16h*)(vbase + (size_t)(3 * 16 + ln) * SEQ + kv + g * 16);

    // prefetch next iteration's K/V into cache (global_prefetch_b8)
    __builtin_prefetch(kbase + (size_t)(kv + 32 + ln) * DH, 0, 0);
    __builtin_prefetch(kbase + (size_t)(kv + 48 + ln) * DH, 0, 0);
#pragma unroll
    for (int t = 0; t < 4; ++t)
      __builtin_prefetch(vbase + (size_t)(t * 16 + ln) * SEQ + kv + 32 + g * 16, 0, 0);

    // ---- per q-tile: scores, online softmax (base-2), P x V ----
#pragma unroll
    for (int qi = 0; qi < 2; ++qi) {
      v16h qf0 = qi ? qb10 : qb00;
      v16h qf1 = qi ? qb11 : qb01;
      float m = qi ? m1 : m0;
      float l = qi ? l1 : l0;

      v8f s0 = {}, s1 = {};
      s0 = wmma16(ka00, qf0, s0);  s0 = wmma16(ka01, qf1, s0);
      s1 = wmma16(ka10, qf0, s1);  s1 = wmma16(ka11, qf1, s1);

      // row stats: rows of P = queries = columns of S^T (col = ln)
      float mloc = s0[0];
#pragma unroll
      for (int r = 1; r < 8; ++r) mloc = fmaxf(mloc, s0[r]);
#pragma unroll
      for (int r = 0; r < 8; ++r) mloc = fmaxf(mloc, s1[r]);
      mloc = fmaxf(mloc, __shfl_xor(mloc, 16, 32));  // join key-halves
      float mnew  = fmaxf(m, mloc);
      float alpha = __builtin_amdgcn_exp2f(m - mnew);
      m = mnew;

      // exp2 -> P directly in the A-operand layout (e<8: tile0, e>=8: tile1)
      v16h pa;
      float rsum = 0.0f;
#pragma unroll
      for (int r = 0; r < 8; ++r) {
        float p0 = __builtin_amdgcn_exp2f(s0[r] - mnew);
        float p1 = __builtin_amdgcn_exp2f(s1[r] - mnew);
        rsum += p0 + p1;
        pa[r]     = (_Float16)p0;
        pa[8 + r] = (_Float16)p1;
      }
      rsum += __shfl_xor(rsum, 16, 32);
      l = l * alpha + rsum;

      if (qi == 0) {
#pragma unroll
        for (int r = 0; r < 8; ++r) {
          float ar = __shfl(alpha, g * 8 + r, 32);  // alpha of query g*8+r
          o00[r] *= ar; o01[r] *= ar; o02[r] *= ar; o03[r] *= ar;
        }
        o00 = wmma16(pa, vb0, o00);
        o01 = wmma16(pa, vb1, o01);
        o02 = wmma16(pa, vb2, o02);
        o03 = wmma16(pa, vb3, o03);
        m0 = m; l0 = l;
      } else {
#pragma unroll
        for (int r = 0; r < 8; ++r) {
          float ar = __shfl(alpha, g * 8 + r, 32);
          o10[r] *= ar; o11[r] *= ar; o12[r] *= ar; o13[r] *= ar;
        }
        o10 = wmma16(pa, vb0, o10);
        o11 = wmma16(pa, vb1, o11);
        o12 = wmma16(pa, vb2, o12);
        o13 = wmma16(pa, vb3, o13);
        m1 = m; l1 = l;
      }
    }
  }

  // ---- normalize by l and store fp32 output [b][s][h*64+d] ----
  float li0 = 1.0f / l0;
  float li1 = 1.0f / l1;
#pragma unroll
  for (int r = 0; r < 8; ++r) {
    float lr0 = __shfl(li0, g * 8 + r, 32);
    float lr1 = __shfl(li1, g * 8 + r, 32);
    int srow0 = qbase + g * 8 + r;
    int srow1 = qbase + 16 + g * 8 + r;
    float* op0 = out + ((size_t)b * SEQ + srow0) * DEMB + h * DH + ln;
    float* op1 = out + ((size_t)b * SEQ + srow1) * DEMB + h * DH + ln;
    op0[0]  = o00[r] * lr0;
    op0[16] = o01[r] * lr0;
    op0[32] = o02[r] * lr0;
    op0[48] = o03[r] * lr0;
    op1[0]  = o10[r] * lr1;
    op1[16] = o11[r] * lr1;
    op1[32] = o12[r] * lr1;
    op1[48] = o13[r] * lr1;
  }
}

// ============================================================================
extern "C" void kernel_launch(void* const* d_in, const int* in_sizes, int n_in,
                              void* d_out, int out_size, void* d_ws, size_t ws_size,
                              hipStream_t stream) {
  // setup_inputs order: k, q, v, Wk, Wq, Wv (all f32)
  const float* k  = (const float*)d_in[0];
  const float* q  = (const float*)d_in[1];
  const float* v  = (const float*)d_in[2];
  const float* Wk = (const float*)d_in[3];
  const float* Wq = (const float*)d_in[4];
  const float* Wv = (const float*)d_in[5];
  float* out = (float*)d_out;

  const size_t per = (size_t)BB * NH * SEQ * DH;  // elems per projected tensor
  _Float16* Qp = (_Float16*)d_ws;
  _Float16* Kp = Qp + per;
  _Float16* Vt = Kp + per;   // 3 * 8 MB = 24 MB of workspace

  // projections: 3 * B * H * (S/16) waves of 32 threads
  {
    int waves = 3 * BB * NH * (SEQ / 16);           // 12288
    int blocks = waves * 32 / 256;                  // 1536
    mha_proj_kernel<<<blocks, 256, 0, stream>>>(q, k, v, Wq, Wk, Wv, Qp, Kp, Vt);
  }
  // attention: B * H * (S/32) waves, 2 q-tiles each
  {
    int waves = BB * NH * (SEQ / 32);               // 2048
    int blocks = waves * 32 / 256;                  // 256
    mha_attn_kernel<<<blocks, 256, 0, stream>>>(Qp, Kp, Vt, out);
  }
}
